// OdorClassifier_14456859918815
// MI455X (gfx1250) — matrix operand order repaired and checked by reading, not compiled, not run on hardware
//
#include <hip/hip_runtime.h>

typedef __bf16 bf16;
typedef __attribute__((ext_vector_type(16))) __bf16 v16bf;
typedef __attribute__((ext_vector_type(8)))  __bf16 v8bf;
typedef __attribute__((ext_vector_type(4)))  __bf16 v4bf;
typedef __attribute__((ext_vector_type(8)))  float  v8f;
typedef __attribute__((ext_vector_type(4)))  float  v4f;

constexpr int kNodes  = 100000;
constexpr int kEdges  = 1600000;
constexpr int kInDim  = 128;
constexpr int kHidden = 256;
constexpr int kOutDim = 138;
constexpr int kGraphs = 2048;

// ------------------------------------------------ CDNA5 async-to-LDS plumbing

#if defined(__has_builtin)
#if __has_builtin(__builtin_amdgcn_global_load_async_to_lds_b128)
#define ASYNC_LDS 1
#endif
#endif
#ifndef ASYNC_LDS
#define ASYNC_LDS 0
#endif

#if defined(__has_builtin) && __has_builtin(__builtin_amdgcn_s_wait_asynccnt)
#define WAIT_ASYNC() __builtin_amdgcn_s_wait_asynccnt(0)
#elif ASYNC_LDS
#define WAIT_ASYNC() asm volatile("s_wait_asynccnt 0x0" ::: "memory")
#else
#define WAIT_ASYNC()
#endif

// Builtin parameter types per hipcc diagnostic: pointer to 16-byte int vector,
// AS1 (printed "__device__") for source, AS3 for the LDS destination.
typedef int v4i_vs __attribute__((vector_size(16)));
typedef __attribute__((address_space(1))) v4i_vs* gptr_b128;
typedef __attribute__((address_space(3))) v4i_vs* lptr_b128;

// copy 16 bytes global -> LDS (async on CDNA5, sync fallback otherwise)
__device__ __forceinline__ void copy16_g2l(const bf16* g, bf16* l) {
#if ASYNC_LDS
  __builtin_amdgcn_global_load_async_to_lds_b128(
      (gptr_b128)(void*)g, (lptr_b128)(void*)l, 0, 0);
#else
  *(v8bf*)l = *(const v8bf*)g;
#endif
}

// ---------------------------------------------------------------- utilities

__global__ __launch_bounds__(256) void zero_f32(float* __restrict__ p, long long n) {
  long long i = (long long)blockIdx.x * blockDim.x + threadIdx.x;
  if (i < n) p[i] = 0.0f;
}

__global__ __launch_bounds__(256) void deg_count(const long long* __restrict__ dst,
                                                 float* __restrict__ deg, int E) {
  int e = blockIdx.x * blockDim.x + threadIdx.x;
  if (e < E) atomicAdd(&deg[dst[e]], 1.0f);
}

__global__ __launch_bounds__(256) void finalize_dis(float* __restrict__ d, int N) {
  int i = blockIdx.x * blockDim.x + threadIdx.x;
  if (i < N) d[i] = rsqrtf(d[i] + 1.0f);
}

__global__ __launch_bounds__(256) void f32_to_bf16(const float* __restrict__ in,
                                                   bf16* __restrict__ out, int n4) {
  int idx = blockIdx.x * blockDim.x + threadIdx.x;
  if (idx >= n4) return;
  v4f v = ((const v4f*)in)[idx];
  v4bf o;
  o[0] = (bf16)v.x; o[1] = (bf16)v.y; o[2] = (bf16)v.z; o[3] = (bf16)v.w;
  ((v4bf*)out)[idx] = o;
}

// Pack fp32 W[K,F] (row-major) into bf16 WMMA B-fragment order:
// fragment t = ko*(F/16)+nb, lane l: n = nb*16 + (l&15), elems i=0..15 -> k = ko*32 + (l>>4)*16 + i
__global__ __launch_bounds__(256) void pack_w(const float* __restrict__ W,
                                              bf16* __restrict__ Wpk, int K, int F) {
  int idx = blockIdx.x * blockDim.x + threadIdx.x;
  if (idx >= K * F) return;
  int i    = idx & 15;
  int lane = (idx >> 4) & 31;
  int t    = idx >> 9;
  int F16  = F >> 4;
  int nb   = t % F16;
  int ko   = t / F16;
  int n = nb * 16 + (lane & 15);
  int k = ko * 32 + (lane >> 4) * 16 + i;
  Wpk[idx] = (bf16)W[(size_t)k * F + n];
}

// ---------------------------------------------------------------- WMMA GEMM
// C[N,F] = A[N,K] (bf16) @ W[K,F] (pre-packed bf16 fragments), fp32 accumulate.
// block = 128 threads = 4 waves; block tile 128 rows x 64 cols; wave tile 32 x 64.
// A tiles double-buffered in LDS via async global->LDS copies.

#define MT 128
#define NT 64
#define KT 32

__global__ __launch_bounds__(128)
void gemm_wmma_bf16(const bf16* __restrict__ A, const bf16* __restrict__ Wpk,
                    float* __restrict__ C, int N, int K, int F) {
  __shared__ bf16 Alds[2][MT][KT + 8];

  const int tid  = threadIdx.x;
  const int lane = tid & 31;
  const int wave = tid >> 5;
  const int row0 = blockIdx.x * MT;
  const int col0 = blockIdx.y * NT;
  const int F16  = F >> 4;
  const int nKT  = K / KT;
  const int khalf = lane >> 4;

  v8f acc[2][4] = {};

  // zero LDS rows belonging to out-of-range rows (tail block only, both buffers)
  for (int c = tid; c < MT; c += 128) {
    if (row0 + c >= N) {
      v8bf z = {};
      #pragma unroll
      for (int j = 0; j < KT; j += 8) {
        *(v8bf*)&Alds[0][c][j] = z;
        *(v8bf*)&Alds[1][c][j] = z;
      }
    }
  }

  // stage one 128x32 bf16 A tile into LDS buffer bufi (16B chunks, 4 per row)
  auto stage = [&](int kt, int bufi) {
    const bf16* base = A + (size_t)row0 * K + kt * KT;
    for (int c = tid; c < MT * 4; c += 128) {
      int rr = c >> 2, q = c & 3;
      if (row0 + rr < N)
        copy16_g2l(base + (size_t)rr * K + q * 8, &Alds[bufi][rr][q * 8]);
    }
  };

  stage(0, 0);

  for (int kt = 0; kt < nKT; ++kt) {
    const int bufi = kt & 1;
    WAIT_ASYNC();        // my async copies for this buffer have landed
    __syncthreads();     // everyone's copies visible; prior reads of other buffer done
    if (kt + 1 < nKT) stage(kt + 1, bufi ^ 1);

    // B fragments straight from packed global (tiny, L2-resident)
    const v16bf* bp = (const v16bf*)Wpk + ((size_t)kt * F16 + (col0 >> 4)) * 32 + lane;
    v16bf bfr0 = bp[0];
    v16bf bfr1 = bp[32];
    v16bf bfr2 = bp[64];
    v16bf bfr3 = bp[96];

    #pragma unroll
    for (int sub = 0; sub < 2; ++sub) {
      // A fragment: lanes 0-15 need K {0..15}, lanes 16-31 need K {16..31}
      const int m = wave * 32 + sub * 16 + (lane & 15);
      v8bf lo = *(const v8bf*)&Alds[bufi][m][khalf * 16];
      v8bf hi = *(const v8bf*)&Alds[bufi][m][khalf * 16 + 8];
      v16bf afrag = {};
      #pragma unroll
      for (int j = 0; j < 8; ++j) { afrag[j] = lo[j]; afrag[8 + j] = hi[j]; }

      acc[sub][0] = __builtin_amdgcn_wmma_f32_16x16x32_bf16(false, afrag, false, bfr0, (short)0, acc[sub][0], false, false);
      acc[sub][1] = __builtin_amdgcn_wmma_f32_16x16x32_bf16(false, afrag, false, bfr1, (short)0, acc[sub][1], false, false);
      acc[sub][2] = __builtin_amdgcn_wmma_f32_16x16x32_bf16(false, afrag, false, bfr2, (short)0, acc[sub][2], false, false);
      acc[sub][3] = __builtin_amdgcn_wmma_f32_16x16x32_bf16(false, afrag, false, bfr3, (short)0, acc[sub][3], false, false);
    }
    __syncthreads();     // reads of this buffer done before it is re-staged
  }

  // store; D layout: lane&15 = n, VGPR r -> m = (lane>>4)*8 + r
  const int gn = col0 + (lane & 15);
  #pragma unroll
  for (int sub = 0; sub < 2; ++sub) {
    const int mb = row0 + wave * 32 + sub * 16 + (lane >> 4) * 8;
    #pragma unroll
    for (int rr = 0; rr < 8; ++rr) {
      int gm = mb + rr;
      if (gm < N) {
        float* cp = C + (size_t)gm * F + gn;
        cp[0]  = acc[sub][0][rr];
        cp[16] = acc[sub][1][rr];
        cp[32] = acc[sub][2][rr];
        cp[48] = acc[sub][3][rr];
      }
    }
  }
}

// ---------------------------------------------------------------- GCN aggregation

// out[i] = HW[i] * dis[i]^2   (self-loop term, also initializes out)
__global__ __launch_bounds__(256)
void self_init(const float* __restrict__ HW, const float* __restrict__ dis,
               float* __restrict__ out, int N) {
  int idx = blockIdx.x * blockDim.x + threadIdx.x;  // over N * (256/4)
  if (idx >= N * (kHidden / 4)) return;
  int row = idx >> 6;
  float d = dis[row];
  float s = d * d;
  v4f hw = ((const v4f*)HW)[idx];
  v4f o;
  o.x = hw.x * s; o.y = hw.y * s; o.z = hw.z * s; o.w = hw.w * s;
  ((v4f*)out)[idx] = o;
}

// out[dst] += HW[src] * (dis[src]*dis[dst]) — 102 MB dest fits in L2, atomics L2-resident
__global__ __launch_bounds__(256)
void edge_scatter(const long long* __restrict__ src, const long long* __restrict__ dst,
                  const float* __restrict__ dis, const float* __restrict__ HW,
                  float* __restrict__ out, int E) {
  long long idx = (long long)blockIdx.x * blockDim.x + threadIdx.x;  // E * 64
  if (idx >= (long long)E * (kHidden / 4)) return;
  int e  = (int)(idx >> 6);
  int f4 = (int)(idx & 63);
  long long s = src[e];
  long long d = dst[e];
  float norm = dis[s] * dis[d];
  v4f mv = ((const v4f*)(HW + s * kHidden))[f4];
  float* op = out + d * kHidden + (size_t)f4 * 4;
  atomicAdd(op + 0, mv.x * norm);
  atomicAdd(op + 1, mv.y * norm);
  atomicAdd(op + 2, mv.z * norm);
  atomicAdd(op + 3, mv.w * norm);
}

// h1 is consumed only by the next GEMM: emit bf16 directly (halves next A traffic)
__global__ __launch_bounds__(256)
void bias_relu_bf16(const float* __restrict__ agg, const float* __restrict__ b,
                    bf16* __restrict__ outb, int N) {
  int idx = blockIdx.x * blockDim.x + threadIdx.x;
  if (idx >= N * kHidden) return;
  int f = idx & (kHidden - 1);
  float v = agg[idx] + b[f];
  outb[idx] = (bf16)(v > 0.0f ? v : 0.0f);
}

__global__ __launch_bounds__(256)
void bias_relu(float* __restrict__ hbuf, const float* __restrict__ b, int N) {
  int idx = blockIdx.x * blockDim.x + threadIdx.x;
  if (idx >= N * kHidden) return;
  int f = idx & (kHidden - 1);
  float v = hbuf[idx] + b[f];
  hbuf[idx] = v > 0.0f ? v : 0.0f;
}

// ---------------------------------------------------------------- pooling + head

__global__ __launch_bounds__(256)
void pool_scatter(const float* __restrict__ hbuf, const long long* __restrict__ batch,
                  float* __restrict__ sums, float* __restrict__ counts, int N) {
  int idx = blockIdx.x * blockDim.x + threadIdx.x;
  if (idx >= N * kHidden) return;
  int node = idx >> 8;
  int f    = idx & (kHidden - 1);
  long long g = batch[node];
  atomicAdd(&sums[g * kHidden + f], hbuf[idx]);
  if (f == 0) atomicAdd(&counts[g], 1.0f);
}

__global__ __launch_bounds__(128)
void logits_kernel(const float* __restrict__ sums, const float* __restrict__ counts,
                   const float* __restrict__ linW, const float* __restrict__ linb,
                   float* __restrict__ out) {
  int idx = blockIdx.x * blockDim.x + threadIdx.x;
  if (idx >= kGraphs * kOutDim) return;
  int g = idx / kOutDim;
  int o = idx - g * kOutDim;
  float inv = 1.0f / fmaxf(counts[g], 1.0f);
  const float* gs = sums + (size_t)g * kHidden;
  float acc = linb[o];
  #pragma unroll 4
  for (int k = 0; k < kHidden; ++k)
    acc = fmaf(gs[k] * inv, linW[(size_t)k * kOutDim + o], acc);
  out[idx] = acc;
}

// ---------------------------------------------------------------- orchestration

extern "C" void kernel_launch(void* const* d_in, const int* in_sizes, int n_in,
                              void* d_out, int out_size, void* d_ws, size_t ws_size,
                              hipStream_t stream) {
  const float*     x     = (const float*)d_in[0];
  const float*     W1    = (const float*)d_in[1];
  const float*     b1    = (const float*)d_in[2];
  const float*     W2    = (const float*)d_in[3];
  const float*     b2    = (const float*)d_in[4];
  const float*     linW  = (const float*)d_in[5];
  const float*     linb  = (const float*)d_in[6];
  const long long* edge  = (const long long*)d_in[7];  // [2, E] int64
  const long long* batch = (const long long*)d_in[8];  // [N] int64
  float* out = (float*)d_out;

  const long long NF = (long long)kNodes * kHidden;

  char* ws = (char*)d_ws;
  size_t off = 0;
  auto take = [&](size_t bytes) -> char* {
    char* p = ws + off;
    off += (bytes + 255) & ~(size_t)255;
    return p;
  };
  float* bufA = (float*)take((size_t)NF * 4);                  // X@W product (fp32)
  float* bufB = (float*)take((size_t)NF * 4);                  // aggregated layer output
  bf16*  xb   = (bf16*)take((size_t)kNodes * kInDim * 2);      // x in bf16
  bf16*  h1b  = (bf16*)take((size_t)NF * 2);                   // relu(h1) in bf16
  float* dis  = (float*)take((size_t)kNodes * 4);              // deg^{-1/2}
  bf16*  W1pk = (bf16*)take((size_t)kInDim * kHidden * 2);
  bf16*  W2pk = (bf16*)take((size_t)kHidden * kHidden * 2);
  float* sums = (float*)take((size_t)kGraphs * kHidden * 4);   // pool sums
  float* cnts = (float*)take((size_t)kGraphs * 4);             // contiguous after sums

  const long long* srcI = edge;
  const long long* dstI = edge + kEdges;

  // degree^{-1/2}
  zero_f32<<<(kNodes + 255) / 256, 256, 0, stream>>>(dis, kNodes);
  deg_count<<<(kEdges + 255) / 256, 256, 0, stream>>>(dstI, dis, kEdges);
  finalize_dis<<<(kNodes + 255) / 256, 256, 0, stream>>>(dis, kNodes);

  // precision prep: weights -> packed bf16 fragments, x -> bf16
  pack_w<<<(kInDim * kHidden + 255) / 256, 256, 0, stream>>>(W1, W1pk, kInDim, kHidden);
  pack_w<<<(kHidden * kHidden + 255) / 256, 256, 0, stream>>>(W2, W2pk, kHidden, kHidden);
  f32_to_bf16<<<((kNodes * kInDim / 4) + 255) / 256, 256, 0, stream>>>(
      x, xb, kNodes * kInDim / 4);

  dim3 ggrid((kNodes + MT - 1) / MT, kHidden / NT);
  int selfBlocks = (kNodes * (kHidden / 4) + 255) / 256;
  int edgeBlocks = (int)(((long long)kEdges * (kHidden / 4) + 255) / 256);
  int nfBlocks   = (int)((NF + 255) / 256);

  // layer 1: bufA = x @ W1 ; h1b = bf16(relu(agg(bufA) + b1))
  gemm_wmma_bf16<<<ggrid, 128, 0, stream>>>(xb, W1pk, bufA, kNodes, kInDim, kHidden);
  self_init<<<selfBlocks, 256, 0, stream>>>(bufA, dis, bufB, kNodes);
  edge_scatter<<<edgeBlocks, 256, 0, stream>>>(srcI, dstI, dis, bufA, bufB, kEdges);
  bias_relu_bf16<<<nfBlocks, 256, 0, stream>>>(bufB, b1, h1b, kNodes);

  // layer 2: bufA = h1 @ W2 ; bufB = relu(agg(bufA) + b2)
  gemm_wmma_bf16<<<ggrid, 128, 0, stream>>>(h1b, W2pk, bufA, kNodes, kHidden, kHidden);
  self_init<<<selfBlocks, 256, 0, stream>>>(bufA, dis, bufB, kNodes);
  edge_scatter<<<edgeBlocks, 256, 0, stream>>>(srcI, dstI, dis, bufA, bufB, kEdges);
  bias_relu<<<nfBlocks, 256, 0, stream>>>(bufB, b2, kNodes);

  // pool (mean per graph) + classifier head
  zero_f32<<<((kGraphs * kHidden + kGraphs) + 255) / 256, 256, 0, stream>>>(
      sums, (long long)kGraphs * kHidden + kGraphs);
  pool_scatter<<<nfBlocks, 256, 0, stream>>>(bufB, batch, sums, cnts, kNodes);
  logits_kernel<<<(kGraphs * kOutDim + 127) / 128, 128, 0, stream>>>(sums, cnts, linW, linb, out);

  (void)in_sizes; (void)n_in; (void)out_size; (void)ws_size;
}
